// MyLstmModel_36275293782284
// MI455X (gfx1250) — matrix operand order repaired and checked by reading, not compiled, not run on hardware
//
#include <hip/hip_runtime.h>

typedef __bf16 bf16_t;
typedef __attribute__((ext_vector_type(16))) __bf16 v16bf;
typedef __attribute__((ext_vector_type(8)))  __bf16 v8bf;
typedef __attribute__((ext_vector_type(8)))  float  v8f;

#define BATCH 1024
#define SEQT  256

// Branch-free activations on v_exp_f32 / v_rcp_f32 (no libm EXEC-mask churn).
__device__ __forceinline__ float fast_rcp(float x) { return __builtin_amdgcn_rcpf(x); }
__device__ __forceinline__ float sigmoidf_(float x) { return fast_rcp(1.0f + __expf(-x)); }
__device__ __forceinline__ float tanhf_(float x) {
  // 1 - 2/(e^{2x}+1): exact saturation at +-1, NaN-free, branch-free.
  return 1.0f - 2.0f * fast_rcp(__expf(2.0f * x) + 1.0f);
}

// Load 8 consecutive elements as bf16 (converting if the source is f32).
__device__ __forceinline__ v8bf load8_bf(const bf16_t* p) { return *(const v8bf*)p; }
__device__ __forceinline__ v8bf load8_bf(const float* p) {
  const float4 a = *(const float4*)p;
  const float4 b = *(const float4*)(p + 4);
  v8bf r;
  r[0] = (bf16_t)a.x; r[1] = (bf16_t)a.y; r[2] = (bf16_t)a.z; r[3] = (bf16_t)a.w;
  r[4] = (bf16_t)b.x; r[5] = (bf16_t)b.y; r[6] = (bf16_t)b.z; r[7] = (bf16_t)b.w;
  return r;
}

// ---------------------------------------------------------------------------
// Swizzle an f32 weight matrix [K x N4] (row-major) into WMMA B-fragment order
// (bf16). Fragment layout per v_wmma_*_16x16x32_bf16 B operand:
//   dst[((kt*NT + nt)*32 + lane)*16 + e] = W[kt*32 + (lane>>4)*8 + (e&7) + (e>>3)*16][nt*16 + (lane&15)]
// Rows k >= K are zero-padded (used for layer-3 U where H=16 < 32).
// ---------------------------------------------------------------------------
__global__ void swizzle_b_kernel(const float* __restrict__ src, bf16_t* __restrict__ dst,
                                 int K, int N4, int KT) {
  int idx = blockIdx.x * blockDim.x + threadIdx.x;
  int NT = N4 >> 4;
  int total = KT * NT * 512;
  if (idx >= total) return;
  int e    = idx & 15;
  int lane = (idx >> 4) & 31;
  int ft   = idx >> 9;          // kt*NT + nt
  int nt   = ft % NT;
  int kt   = ft / NT;
  int n = nt * 16 + (lane & 15);
  int k = kt * 32 + ((lane >> 4) << 3) + (e & 7) + ((e >> 3) << 4);
  float v = (k < K) ? src[(size_t)k * N4 + n] : 0.0f;
  dst[idx] = (bf16_t)v;
}

// ---------------------------------------------------------------------------
// Fused bidirectional LSTM layer scan.
//   grid  = (BATCH/16, 2)    blockIdx.y = direction (0 fwd, 1 bwd)
//   block = 32 * NH threads  (NH = H/16 waves; wave w owns h columns w*16..w*16+15
//           and computes gate N-tiles {w, w+NH, w+2NH, w+3NH} = its i,f,g,o)
// Weight B-fragments live in VGPRs for the entire T loop; x A-fragments are
// loaded straight from global; only the h transpose goes through LDS.
// ---------------------------------------------------------------------------
template <int IN, int H, typename TI, bool WRITE_SEQ>
__global__ __launch_bounds__(32 * (H / 16))
void lstm_bidir_kernel(const TI* __restrict__ xin,
                       const bf16_t* __restrict__ wF, const bf16_t* __restrict__ uF,
                       const float* __restrict__ bF,
                       const bf16_t* __restrict__ wB, const bf16_t* __restrict__ uB,
                       const float* __restrict__ bB,
                       bf16_t* __restrict__ outseq,   // [B, T, 2H] bf16 (WRITE_SEQ)
                       float* __restrict__ lasth) {   // [B, 2H] f32 (!WRITE_SEQ)
  constexpr int KTX = IN / 32;                 // K-tiles for x@W
  constexpr int KPU = (H < 32) ? 32 : H;       // padded K for h@U
  constexpr int KTU = KPU / 32;                // K-tiles for h@U
  constexpr int NT  = (4 * H) / 16;            // N-tiles across the 4H gate width
  constexpr int NH  = H / 16;                  // waves per block

  __shared__ alignas(32) bf16_t sH[16 * KPU];

  const int tid   = threadIdx.x;
  const int lane  = tid & 31;
  const int wv    = tid >> 5;                  // wave id: 0..NH-1
  const int dir   = blockIdx.y;
  const int b0    = blockIdx.x * 16;
  const int mhalf = lane >> 4;
  const int ncol  = lane & 15;
  const int mrow  = lane & 15;

  const bf16_t* __restrict__ wswz = dir ? wB : wF;
  const bf16_t* __restrict__ uswz = dir ? uB : uF;
  const float*  __restrict__ bias = dir ? bB : bF;
  const int col_off = dir * H;

  // ---- preload weight B-fragments into VGPRs (loop-invariant over T) -------
  v16bf wfrag[4 * KTX];
  v16bf ufrag[4 * KTU];
#pragma unroll
  for (int g = 0; g < 4; ++g) {
    const int nt = g * NH + wv;
#pragma unroll
    for (int kt = 0; kt < KTX; ++kt)
      wfrag[g * KTX + kt] = *(const v16bf*)&wswz[((kt * NT + nt) * 32 + lane) * 16];
#pragma unroll
    for (int kt = 0; kt < KTU; ++kt)
      ufrag[g * KTU + kt] = *(const v16bf*)&uswz[((kt * NT + nt) * 32 + lane) * 16];
  }

  float bias_r[4];
#pragma unroll
  for (int g = 0; g < 4; ++g) bias_r[g] = bias[(g * NH + wv) * 16 + ncol];

  v8f c;
#pragma unroll
  for (int r = 0; r < 8; ++r) c[r] = 0.0f;

  // h staging starts at zero (also provides the K-padding zeros for H<32).
  for (int i = tid; i < 16 * KPU; i += 32 * NH) sH[i] = (bf16_t)0.0f;
  __syncthreads();

  for (int tt = 0; tt < SEQT; ++tt) {
    const int t = dir ? (SEQT - 1 - tt) : tt;

    // ---- A fragments for x_t, straight from global --------------------------
    // ISA 16-bit A layout: lane L -> row M=L&15; elems 0..7 = K kb..kb+7,
    // elems 8..15 = K kb+16..kb+23, kb = kt*32 + (L>>4)*8.
    v16bf afx[KTX];
#pragma unroll
    for (int kt = 0; kt < KTX; ++kt) {
      const TI* row = xin + ((size_t)(b0 + mrow) * SEQT + t) * IN + kt * 32 + (mhalf << 3);
      union { v16bf v; struct { v8bf lo, hi; } s; } u;
      u.s.lo = load8_bf(row);
      u.s.hi = load8_bf(row + 16);
      afx[kt] = u.v;
    }
    // ---- A fragments for h_{t-1}, from LDS ----------------------------------
    v16bf afh[KTU];
#pragma unroll
    for (int kt = 0; kt < KTU; ++kt) {
      const bf16_t* row = &sH[mrow * KPU + kt * 32 + (mhalf << 3)];
      union { v16bf v; struct { v8bf lo, hi; } s; } u;
      u.s.lo = *(const v8bf*)row;
      u.s.hi = *(const v8bf*)(row + 16);
      afh[kt] = u.v;
    }
    __syncthreads();   // everyone's sH reads complete before anyone rewrites it

    // ---- z tiles for this wave's i,f,g,o ------------------------------------
    v8f acc[4];
#pragma unroll
    for (int g = 0; g < 4; ++g) {
      v8f a;
#pragma unroll
      for (int r = 0; r < 8; ++r) a[r] = bias_r[g];
#pragma unroll
      for (int kt = 0; kt < KTX; ++kt)
        a = __builtin_amdgcn_wmma_f32_16x16x32_bf16(false, afx[kt], false,
                                                    wfrag[g * KTX + kt],
                                                    (short)0, a, false, false);
#pragma unroll
      for (int kt = 0; kt < KTU; ++kt)
        a = __builtin_amdgcn_wmma_f32_16x16x32_bf16(false, afh[kt], false,
                                                    ufrag[g * KTU + kt],
                                                    (short)0, a, false, false);
      acc[g] = a;
    }

    // ---- gates + state update; C layout: M = r + 8*(L>=16), N = L&15 --------
#pragma unroll
    for (int r = 0; r < 8; ++r) {
      float ig = sigmoidf_(acc[0][r]);
      float fg = sigmoidf_(acc[1][r]);
      float gg = tanhf_(acc[2][r]);
      float og = sigmoidf_(acc[3][r]);
      float cc = fg * c[r] + ig * gg;
      c[r] = cc;
      float hh = og * tanhf_(cc);
      bf16_t hb = (bf16_t)hh;
      const int m = r + (mhalf << 3);
      const int n = wv * 16 + ncol;
      sH[m * KPU + n] = hb;
      if constexpr (WRITE_SEQ) {
        outseq[((size_t)(b0 + m) * SEQT + t) * (2 * H) + col_off + n] = hb;
      }
    }
    __syncthreads();   // new h visible before next step's reads
  }

  if constexpr (!WRITE_SEQ) {
    for (int i = tid; i < 16 * H; i += 32 * NH) {
      int m = i / H, n = i % H;
      lasth[(size_t)(b0 + m) * (2 * H) + col_off + n] = (float)sH[m * KPU + n];
    }
  }
}

// ---------------------------------------------------------------------------
// Final head: relu6(h[B,32] @ Wd1[32,16] + bd1) @ Wd2[16,2] + bd2 -> softmax.
// ---------------------------------------------------------------------------
__global__ void dense_softmax_kernel(const float* __restrict__ h,
                                     const float* __restrict__ Wd1,
                                     const float* __restrict__ bd1,
                                     const float* __restrict__ Wd2,
                                     const float* __restrict__ bd2,
                                     float* __restrict__ out) {
  int b = blockIdx.x * blockDim.x + threadIdx.x;
  if (b >= BATCH) return;
  float hv[32];
#pragma unroll
  for (int k = 0; k < 32; ++k) hv[k] = h[b * 32 + k];
  float l0 = bd2[0], l1 = bd2[1];
#pragma unroll
  for (int j = 0; j < 16; ++j) {
    float s = bd1[j];
#pragma unroll
    for (int k = 0; k < 32; ++k) s += hv[k] * Wd1[k * 16 + j];
    s = fminf(fmaxf(s, 0.0f), 6.0f);
    l0 += s * Wd2[j * 2 + 0];
    l1 += s * Wd2[j * 2 + 1];
  }
  float mx = fmaxf(l0, l1);
  float e0 = __expf(l0 - mx), e1 = __expf(l1 - mx);
  float inv = fast_rcp(e0 + e1);
  out[b * 2 + 0] = e0 * inv;
  out[b * 2 + 1] = e1 * inv;
}

// ---------------------------------------------------------------------------
extern "C" void kernel_launch(void* const* d_in, const int* in_sizes, int n_in,
                              void* d_out, int out_size, void* d_ws, size_t ws_size,
                              hipStream_t stream) {
  (void)in_sizes; (void)n_in; (void)out_size; (void)ws_size;

  const float* x   = (const float*)d_in[0];
  const float* W1f = (const float*)d_in[1];
  const float* U1f = (const float*)d_in[2];
  const float* b1f = (const float*)d_in[3];
  const float* W1b = (const float*)d_in[4];
  const float* U1b = (const float*)d_in[5];
  const float* b1b = (const float*)d_in[6];
  const float* W2f = (const float*)d_in[7];
  const float* U2f = (const float*)d_in[8];
  const float* b2f = (const float*)d_in[9];
  const float* W2b = (const float*)d_in[10];
  const float* U2b = (const float*)d_in[11];
  const float* b2b = (const float*)d_in[12];
  const float* W3f = (const float*)d_in[13];
  const float* U3f = (const float*)d_in[14];
  const float* b3f = (const float*)d_in[15];
  const float* W3b = (const float*)d_in[16];
  const float* U3b = (const float*)d_in[17];
  const float* b3b = (const float*)d_in[18];
  const float* Wd1 = (const float*)d_in[19];
  const float* bd1 = (const float*)d_in[20];
  const float* Wd2 = (const float*)d_in[21];
  const float* bd2 = (const float*)d_in[22];

  // ---- carve workspace ----------------------------------------------------
  char* p = (char*)d_ws;
  auto carve = [&](size_t bytes) -> char* {
    char* r = p;
    p += (bytes + 255) & ~(size_t)255;
    return r;
  };
  bf16_t* sw1f = (bf16_t*)carve(2 * 16 * 512 * 2);
  bf16_t* su1f = (bf16_t*)carve(2 * 16 * 512 * 2);
  bf16_t* sw1b = (bf16_t*)carve(2 * 16 * 512 * 2);
  bf16_t* su1b = (bf16_t*)carve(2 * 16 * 512 * 2);
  bf16_t* sw2f = (bf16_t*)carve(4 * 8 * 512 * 2);
  bf16_t* su2f = (bf16_t*)carve(1 * 8 * 512 * 2);
  bf16_t* sw2b = (bf16_t*)carve(4 * 8 * 512 * 2);
  bf16_t* su2b = (bf16_t*)carve(1 * 8 * 512 * 2);
  bf16_t* sw3f = (bf16_t*)carve(2 * 4 * 512 * 2);
  bf16_t* su3f = (bf16_t*)carve(1 * 4 * 512 * 2);
  bf16_t* sw3b = (bf16_t*)carve(2 * 4 * 512 * 2);
  bf16_t* su3b = (bf16_t*)carve(1 * 4 * 512 * 2);
  bf16_t* out1 = (bf16_t*)carve((size_t)BATCH * SEQT * 128 * 2);  // 67 MB
  bf16_t* out2 = (bf16_t*)carve((size_t)BATCH * SEQT * 64 * 2);   // 33.5 MB
  float*  h3   = (float*)carve((size_t)BATCH * 32 * 4);

  // ---- weight swizzle into WMMA B-fragment order --------------------------
  auto swz = [&](const float* src, bf16_t* dst, int K, int N4, int KT) {
    int total = KT * (N4 >> 4) * 512;
    swizzle_b_kernel<<<dim3((total + 255) / 256), dim3(256), 0, stream>>>(src, dst, K, N4, KT);
  };
  swz(W1f, sw1f, 64, 256, 2);  swz(U1f, su1f, 64, 256, 2);
  swz(W1b, sw1b, 64, 256, 2);  swz(U1b, su1b, 64, 256, 2);
  swz(W2f, sw2f, 128, 128, 4); swz(U2f, su2f, 32, 128, 1);
  swz(W2b, sw2b, 128, 128, 4); swz(U2b, su2b, 32, 128, 1);
  swz(W3f, sw3f, 64, 64, 2);   swz(U3f, su3f, 16, 64, 1);
  swz(W3b, sw3b, 64, 64, 2);   swz(U3b, su3b, 16, 64, 1);

  // ---- LSTM stack (both directions per launch; stream order = layer deps) --
  dim3 grid(BATCH / 16, 2);
  lstm_bidir_kernel<64, 64, float, true><<<grid, dim3(32 * 4), 0, stream>>>(
      x, sw1f, su1f, b1f, sw1b, su1b, b1b, out1, nullptr);
  lstm_bidir_kernel<128, 32, bf16_t, true><<<grid, dim3(32 * 2), 0, stream>>>(
      out1, sw2f, su2f, b2f, sw2b, su2b, b2b, out2, nullptr);
  lstm_bidir_kernel<64, 16, bf16_t, false><<<grid, dim3(32 * 1), 0, stream>>>(
      out2, sw3f, su3f, b3f, sw3b, su3b, b3b, nullptr, h3);

  // ---- head ----------------------------------------------------------------
  dense_softmax_kernel<<<dim3(BATCH / 256), dim3(256), 0, stream>>>(
      h3, Wd1, bd1, Wd2, bd2, (float*)d_out);
}
